// TransformerLayer_4973572128772
// MI455X (gfx1250) — compile-verified, matrix operand order checked by reading, not run on hardware
//
#include <hip/hip_runtime.h>
#include <cstddef>

// ---------------------------------------------------------------------------
// Types for CDNA5 WMMA (wave32): v_wmma_f32_16x16x32_bf16
// ---------------------------------------------------------------------------
typedef __bf16 bf16;
typedef __attribute__((ext_vector_type(16))) __bf16 v16bf;
typedef __attribute__((ext_vector_type(8)))  __bf16 v8bf;
typedef __attribute__((ext_vector_type(8)))  float  v8f;

#define WMMA_BF16(a, b, c) \
  __builtin_amdgcn_wmma_f32_16x16x32_bf16(false, (a), false, (b), (short)0, (c), false, false)

static constexpr int Bd   = 4;
static constexpr int Sd   = 1024;
static constexpr int Dd   = 768;
static constexpr int Hd   = 12;
static constexpr int DHd  = 64;
static constexpr int Ed   = 8;
static constexpr int HIDd = 3072;
static constexpr int Mrows = Bd * Sd;   // 4096

// ---------------------------------------------------------------------------
// LayerNorm over D=768 per row.  in (+ optional add) -> LN -> *g + b
// optional bf16 out, optional f32 out, optional extra residual added to f32 out
// ---------------------------------------------------------------------------
__global__ __launch_bounds__(256)
void ln_kernel(const float* __restrict__ in, const float* __restrict__ add,
               const float* __restrict__ g,  const float* __restrict__ bb,
               bf16* __restrict__ outB, float* __restrict__ outF,
               const float* __restrict__ res)
{
  const int row = blockIdx.x;
  const float* x = in + (size_t)row * Dd;
  __shared__ float red[256];
  float v0[3];
  float s = 0.f;
#pragma unroll
  for (int i = 0; i < 3; ++i) {
    int c = threadIdx.x + 256 * i;
    float t = x[c];
    if (add) t += add[(size_t)row * Dd + c];
    v0[i] = t; s += t;
  }
  red[threadIdx.x] = s; __syncthreads();
  for (int o = 128; o > 0; o >>= 1) {
    if (threadIdx.x < (unsigned)o) red[threadIdx.x] += red[threadIdx.x + o];
    __syncthreads();
  }
  const float mean = red[0] * (1.0f / Dd);
  __syncthreads();
  float s2 = 0.f;
#pragma unroll
  for (int i = 0; i < 3; ++i) { float d = v0[i] - mean; s2 += d * d; }
  red[threadIdx.x] = s2; __syncthreads();
  for (int o = 128; o > 0; o >>= 1) {
    if (threadIdx.x < (unsigned)o) red[threadIdx.x] += red[threadIdx.x + o];
    __syncthreads();
  }
  const float rstd = rsqrtf(red[0] * (1.0f / Dd) + 1e-5f);
#pragma unroll
  for (int i = 0; i < 3; ++i) {
    int c = threadIdx.x + 256 * i;
    float y = (v0[i] - mean) * rstd * g[c] + bb[c];
    if (outB) outB[(size_t)row * Dd + c] = (bf16)y;
    if (outF) {
      float yo = y;
      if (res) yo += res[(size_t)row * Dd + c];
      outF[(size_t)row * Dd + c] = yo;
    }
  }
}

// ---------------------------------------------------------------------------
// Gate: logits = h @ gate_w + gate_b ; top-2 softmax -> per-token scale[E]
// ---------------------------------------------------------------------------
__global__ __launch_bounds__(256)
void gate_kernel(const float* __restrict__ h, const float* __restrict__ gw,
                 const float* __restrict__ gb, float* __restrict__ scale)
{
  const int m = blockIdx.x * 256 + threadIdx.x;
  const float* hr = h + (size_t)m * Dd;
  float lg[Ed];
#pragma unroll
  for (int e = 0; e < Ed; ++e) lg[e] = gb[e];
  for (int d = 0; d < Dd; ++d) {
    const float hv = hr[d];
#pragma unroll
    for (int e = 0; e < Ed; ++e) lg[e] += hv * gw[d * Ed + e];
  }
  int i1 = 0;
#pragma unroll
  for (int e = 1; e < Ed; ++e) if (lg[e] > lg[i1]) i1 = e;
  int i2 = (i1 == 0) ? 1 : 0;
#pragma unroll
  for (int e = 0; e < Ed; ++e) if (e != i1 && lg[e] > lg[i2]) i2 = e;
  const float v1 = lg[i1], v2 = lg[i2];
  const float e1 = 1.0f, e2 = __expf(v2 - v1);
  const float inv = 1.0f / (e1 + e2);
  float* sr = scale + (size_t)m * Ed;
#pragma unroll
  for (int e = 0; e < Ed; ++e) sr[e] = 0.f;
  sr[i1] = e1 * inv;
  sr[i2] = e2 * inv;
}

// ---------------------------------------------------------------------------
// Tiled WMMA GEMM:  C[M,N] = A(bf16,[M,K]) * Bw(f32,[K,N]) + bias
// macro tile 128x128, 8 waves (4x2) of 32x64, K step 32. Fused epilogues.
// Per wave per K-step: 8 WMMA vs 12 ds_load_b128 fragment loads.
// ---------------------------------------------------------------------------
enum { EPI_QKV = 0, EPI_WO = 1, EPI_RELU = 2, EPI_SCALE_ACC = 3 };

template <int EPI>
__global__ __launch_bounds__(256)
void gemm_wmma(const bf16* __restrict__ A, const float* __restrict__ Bw,
               const float* __restrict__ bias, const float* __restrict__ res,
               const float* __restrict__ scaleRow,  // stride Ed per row
               float* __restrict__ outF, bf16* __restrict__ outB,
               int M, int N, int K, int accum)
{
  __shared__ __align__(16) bf16 As[128][40];   // [m][k], padded stride 40
  __shared__ __align__(16) bf16 Bs[128][40];   // [n][k] (transposed), padded

  const int bm = blockIdx.y * 128, bn = blockIdx.x * 128;
  const int tid = threadIdx.x;
  const int lane = tid & 31, wv = tid >> 5;
  const int wm = (wv & 3) * 32, wn = (wv >> 2) * 64;
  const int ln16 = lane & 15, hi = lane >> 4;

  v8f zero8;
#pragma unroll
  for (int j = 0; j < 8; ++j) zero8[j] = 0.f;
  v8f acc[2][4];
#pragma unroll
  for (int i = 0; i < 2; ++i)
#pragma unroll
    for (int j = 0; j < 4; ++j) acc[i][j] = zero8;

  for (int kb = 0; kb < K; kb += 32) {
    // ---- stage A tile (128x32 bf16): two b128 per thread
#pragma unroll
    for (int i = 0; i < 2; ++i) {
      const int idx = tid * 8 + i * 2048;
      const int r = idx >> 5, c = idx & 31;
      *(v8bf*)&As[r][c] = *(const v8bf*)(A + (size_t)(bm + r) * K + kb + c);
    }
    // ---- stage B tile (32x128 f32 -> bf16, transposed into [n][k])
#pragma unroll
    for (int i = 0; i < 4; ++i) {
      const int idx = tid * 4 + i * 1024;
      const int kr = idx >> 7, nc = idx & 127;
      const float4 f = *(const float4*)(Bw + (size_t)(kb + kr) * N + bn + nc);
      Bs[nc + 0][kr] = (bf16)f.x;
      Bs[nc + 1][kr] = (bf16)f.y;
      Bs[nc + 2][kr] = (bf16)f.z;
      Bs[nc + 3][kr] = (bf16)f.w;
    }
    if (kb + 32 < K) {
      __builtin_prefetch(A  + (size_t)(bm + (tid >> 2)) * K + kb + 32, 0, 1);
      __builtin_prefetch(Bw + (size_t)(kb + 32 + (tid >> 3)) * N + bn, 0, 1);
    }
    __syncthreads();

    // ---- fragments (ISA 16-bit A layout: lane<16: K 0..7,16..23; lane>=16: 8..15,24..31)
    v16bf af[2], bfm[4];
#pragma unroll
    for (int i = 0; i < 2; ++i) {
      const bf16* ap = &As[wm + i * 16 + ln16][0];
      v8bf lo = *(const v8bf*)(ap + 8 * hi);
      v8bf h8 = *(const v8bf*)(ap + 16 + 8 * hi);
#pragma unroll
      for (int j = 0; j < 8; ++j) { af[i][j] = lo[j]; af[i][j + 8] = h8[j]; }
    }
#pragma unroll
    for (int i = 0; i < 4; ++i) {
      const bf16* bp = &Bs[wn + i * 16 + ln16][0];
      v8bf bl = *(const v8bf*)(bp + 8 * hi);
      v8bf bh = *(const v8bf*)(bp + 16 + 8 * hi);
#pragma unroll
      for (int j = 0; j < 8; ++j) { bfm[i][j] = bl[j]; bfm[i][j + 8] = bh[j]; }
    }
#pragma unroll
    for (int i = 0; i < 2; ++i)
#pragma unroll
      for (int j = 0; j < 4; ++j)
        acc[i][j] = WMMA_BF16(af[i], bfm[j], acc[i][j]);
    __syncthreads();
  }

  // ---- epilogue (C layout: N = lane&15, M = elem + 8*(lane>=16))
#pragma unroll
  for (int i = 0; i < 2; ++i)
#pragma unroll
    for (int j = 0; j < 4; ++j) {
#pragma unroll
      for (int e = 0; e < 8; ++e) {
        const int gm = bm + wm + i * 16 + e + 8 * hi;
        const int gn = bn + wn + j * 16 + ln16;
        const float val = acc[i][j][e] + bias[gn];
        if (EPI == EPI_QKV) {
          // scatter [M=b*S+s, n=h*64+d] -> [B,H,S,DH] bf16
          const int b = gm >> 10, s = gm & 1023, hh = gn >> 6, d = gn & 63;
          outB[(((size_t)(b * Hd + hh)) * Sd + s) * DHd + d] = (bf16)val;
        } else if (EPI == EPI_WO) {
          const size_t o = (size_t)gm * N + gn;
          outF[o] = val + res[o];
        } else if (EPI == EPI_RELU) {
          outB[(size_t)gm * N + gn] = (bf16)fmaxf(val, 0.f);
        } else {  // EPI_SCALE_ACC
          const float sc = scaleRow[(size_t)gm * Ed];
          const size_t o = (size_t)gm * N + gn;
          const float prev = accum ? outF[o] : 0.f;
          outF[o] = prev + sc * val;
        }
      }
    }
}

// ---------------------------------------------------------------------------
// Flash attention: block = (64 q rows, one b*H+h), 4 waves x 16 q rows.
// Online softmax, causal mask, 32-key tiles, WMMA for QK^T and PV.
// ---------------------------------------------------------------------------
__global__ __launch_bounds__(128)
void attn_kernel(const bf16* __restrict__ Q, const bf16* __restrict__ Kp,
                 const bf16* __restrict__ Vp, bf16* __restrict__ ctx)
{
  __shared__ __align__(16) bf16 Vt[64][40];        // V^T tile: [d][k]
  __shared__ __align__(16) bf16 Pt[4][16][40];     // per-wave P tile [m][k]

  const int tid = threadIdx.x, lane = tid & 31, wv = tid >> 5;
  const int ln16 = lane & 15, hi = lane >> 4;
  const int bh = blockIdx.y;                 // b*H + h
  const int b = bh / Hd, h = bh % Hd;
  const int qr0 = blockIdx.x * 64 + wv * 16;
  const bf16* qbase = Q  + (size_t)bh * Sd * DHd;
  const bf16* kbase = Kp + (size_t)bh * Sd * DHd;
  const bf16* vbase = Vp + (size_t)bh * Sd * DHd;

  v8f zero8;
#pragma unroll
  for (int j = 0; j < 8; ++j) zero8[j] = 0.f;

  // Q A-fragments (16x64 = 2 fragments over k-dim halves)
  v16bf qa[2];
#pragma unroll
  for (int f = 0; f < 2; ++f) {
    const bf16* qp = qbase + (size_t)(qr0 + ln16) * DHd + f * 32 + 8 * hi;
    v8bf lo = *(const v8bf*)qp;
    v8bf h8 = *(const v8bf*)(qp + 16);
#pragma unroll
    for (int j = 0; j < 8; ++j) { qa[f][j] = lo[j]; qa[f][j + 8] = h8[j]; }
  }

  v8f o[4];
#pragma unroll
  for (int i = 0; i < 4; ++i) o[i] = zero8;
  float mrow[8], lrow[8];
#pragma unroll
  for (int i = 0; i < 8; ++i) { mrow[i] = -1e30f; lrow[i] = 0.f; }

  const int NT = blockIdx.x * 2 + 2;   // uniform across block; extra tiles fully masked
  for (int t = 0; t < NT; ++t) {
    const int k0 = t * 32;
    // stage V tile transposed (all 128 threads)
#pragma unroll
    for (int i = 0; i < 16; ++i) {
      const int idx = i * 128 + tid;
      const int kk = idx >> 6, d = idx & 63;
      Vt[d][kk] = vbase[(size_t)(k0 + kk) * DHd + d];
    }
    __syncthreads();

    // scores: S = Q (16x64) * K^T (64x32) -> two 16x16 C tiles
    v8f sc[2];
#pragma unroll
    for (int sn = 0; sn < 2; ++sn) {
      v8f c = zero8;
#pragma unroll
      for (int f = 0; f < 2; ++f) {
        const bf16* kp = kbase + (size_t)(k0 + sn * 16 + ln16) * DHd + f * 32 + 8 * hi;
        v8bf lo = *(const v8bf*)kp;
        v8bf h8 = *(const v8bf*)(kp + 16);
        v16bf kb;
#pragma unroll
        for (int j = 0; j < 8; ++j) { kb[j] = lo[j]; kb[j + 8] = h8[j]; }
        c = WMMA_BF16(qa[f], kb, c);
      }
      sc[sn] = c;
    }

    // causal mask + online softmax (row reduce over 16-lane group)
    float p[2][8];
#pragma unroll
    for (int e = 0; e < 8; ++e) {
      const int qg = qr0 + e + 8 * hi;
      float v0 = sc[0][e] * 0.125f;     // 1/sqrt(64)
      float v1 = sc[1][e] * 0.125f;
      if (k0 + ln16 > qg)      v0 = -1e30f;
      if (k0 + 16 + ln16 > qg) v1 = -1e30f;
      float rmax = fmaxf(v0, v1);
      for (int ms = 1; ms < 16; ms <<= 1) rmax = fmaxf(rmax, __shfl_xor(rmax, ms, 32));
      const float nm = fmaxf(mrow[e], rmax);
      const float al = __expf(mrow[e] - nm);
      const float p0 = __expf(v0 - nm), p1 = __expf(v1 - nm);
      float rs = p0 + p1;
      for (int ms = 1; ms < 16; ms <<= 1) rs += __shfl_xor(rs, ms, 32);
      lrow[e] = lrow[e] * al + rs;
      mrow[e] = nm;
      p[0][e] = p0; p[1][e] = p1;
#pragma unroll
      for (int dt = 0; dt < 4; ++dt) o[dt][e] *= al;
    }

    // C-layout P -> row-major LDS -> A-fragment
#pragma unroll
    for (int sn = 0; sn < 2; ++sn)
#pragma unroll
      for (int e = 0; e < 8; ++e)
        Pt[wv][e + 8 * hi][sn * 16 + ln16] = (bf16)p[sn][e];
    __syncthreads();

    v16bf pa;
    {
      const bf16* pp = &Pt[wv][ln16][0];
      v8bf lo = *(const v8bf*)(pp + 8 * hi);
      v8bf h8 = *(const v8bf*)(pp + 16 + 8 * hi);
#pragma unroll
      for (int j = 0; j < 8; ++j) { pa[j] = lo[j]; pa[j + 8] = h8[j]; }
    }

    // O += P (16x32) * V (32x64): 4 d-tiles
#pragma unroll
    for (int dt = 0; dt < 4; ++dt) {
      const bf16* vp = &Vt[dt * 16 + ln16][0];
      v8bf lo = *(const v8bf*)(vp + 8 * hi);
      v8bf h8 = *(const v8bf*)(vp + 16 + 8 * hi);
      v16bf vb;
#pragma unroll
      for (int j = 0; j < 8; ++j) { vb[j] = lo[j]; vb[j + 8] = h8[j]; }
      o[dt] = WMMA_BF16(pa, vb, o[dt]);
    }
    __syncthreads();
  }

  // normalize + store ctx [B,S,D] bf16
#pragma unroll
  for (int e = 0; e < 8; ++e) {
    const float inv = 1.0f / lrow[e];
    const int srow = qr0 + e + 8 * hi;
#pragma unroll
    for (int dt = 0; dt < 4; ++dt) {
      const float val = o[dt][e] * inv;
      ctx[((size_t)(b * Sd + srow)) * Dd + h * DHd + dt * 16 + ln16] = (bf16)val;
    }
  }
}

// ---------------------------------------------------------------------------
// Host orchestration
// ---------------------------------------------------------------------------
extern "C" void kernel_launch(void* const* d_in, const int* in_sizes, int n_in,
                              void* d_out, int out_size, void* d_ws, size_t ws_size,
                              hipStream_t stream)
{
  (void)in_sizes; (void)n_in; (void)out_size; (void)ws_size;
  const float* x   = (const float*)d_in[0];
  // d_in[1] = mask (causal; computed analytically), skipped
  const float* lag = (const float*)d_in[2];
  const float* lab = (const float*)d_in[3];
  const float* wq  = (const float*)d_in[4];
  const float* bq  = (const float*)d_in[5];
  const float* wk  = (const float*)d_in[6];
  const float* bk  = (const float*)d_in[7];
  const float* wvv = (const float*)d_in[8];
  const float* bv  = (const float*)d_in[9];
  const float* wo  = (const float*)d_in[10];
  const float* bo  = (const float*)d_in[11];
  const float* lfg = (const float*)d_in[12];
  const float* lfb = (const float*)d_in[13];
  const float* gw  = (const float*)d_in[14];
  const float* gb  = (const float*)d_in[15];
  const float* w1  = (const float*)d_in[16];
  const float* b1  = (const float*)d_in[17];
  const float* w2  = (const float*)d_in[18];
  const float* b2  = (const float*)d_in[19];
  const float* mlg = (const float*)d_in[20];
  const float* mlb = (const float*)d_in[21];
  // d_in[22] = top_k (hardcoded 2)

  char* ws = (char*)d_ws;
  size_t off = 0;
  auto alloc = [&](size_t bytes) -> char* {
    char* p = ws + off;
    off += (bytes + 255) & ~size_t(255);
    return p;
  };
  bf16*  a_bf   = (bf16*) alloc((size_t)Mrows * Dd * 2);
  bf16*  q_bf   = (bf16*) alloc((size_t)Mrows * Dd * 2);
  bf16*  k_bf   = (bf16*) alloc((size_t)Mrows * Dd * 2);
  bf16*  v_bf   = (bf16*) alloc((size_t)Mrows * Dd * 2);
  bf16*  ctx_bf = (bf16*) alloc((size_t)Mrows * Dd * 2);
  bf16*  h_bf   = (bf16*) alloc((size_t)Mrows * Dd * 2);
  bf16*  t1_bf  = (bf16*) alloc((size_t)Mrows * HIDd * 2);
  float* x1     = (float*)alloc((size_t)Mrows * Dd * 4);
  float* h_f    = (float*)alloc((size_t)Mrows * Dd * 4);
  float* core   = (float*)alloc((size_t)Mrows * Dd * 4);
  float* scl    = (float*)alloc((size_t)Mrows * Ed * 4);

  const dim3 gD(Dd / 128, Mrows / 128);     // 6 x 32
  const dim3 gH(HIDd / 128, Mrows / 128);   // 24 x 32

  // 1) pre-attention LN -> bf16
  ln_kernel<<<Mrows, 256, 0, stream>>>(x, nullptr, lag, lab, a_bf, nullptr, nullptr);
  // 2) Q,K,V projections (WMMA, head-major scatter)
  gemm_wmma<EPI_QKV><<<gD, 256, 0, stream>>>(a_bf, wq,  bq, nullptr, nullptr, nullptr, q_bf, Mrows, Dd, Dd, 0);
  gemm_wmma<EPI_QKV><<<gD, 256, 0, stream>>>(a_bf, wk,  bk, nullptr, nullptr, nullptr, k_bf, Mrows, Dd, Dd, 0);
  gemm_wmma<EPI_QKV><<<gD, 256, 0, stream>>>(a_bf, wvv, bv, nullptr, nullptr, nullptr, v_bf, Mrows, Dd, Dd, 0);
  // 3) flash attention (WMMA)
  attn_kernel<<<dim3(Sd / 64, Bd * Hd), 128, 0, stream>>>(q_bf, k_bf, v_bf, ctx_bf);
  // 4) output projection + residual -> x1 (f32)
  gemm_wmma<EPI_WO><<<gD, 256, 0, stream>>>(ctx_bf, wo, bo, x, nullptr, x1, nullptr, Mrows, Dd, Dd, 0);
  // 5) FF LN -> h (bf16 for GEMMs, f32 for gate/final)
  ln_kernel<<<Mrows, 256, 0, stream>>>(x1, nullptr, lfg, lfb, h_bf, h_f, nullptr);
  // 6) gate top-2 -> per-token expert scales
  gate_kernel<<<Mrows / 256, 256, 0, stream>>>(h_f, gw, gb, scl);
  // 7) experts: FC1+ReLU then FC2 scaled-accumulate into core
  for (int e = 0; e < Ed; ++e) {
    gemm_wmma<EPI_RELU><<<gH, 256, 0, stream>>>(
        h_bf, w1 + (size_t)e * Dd * HIDd, b1 + (size_t)e * HIDd,
        nullptr, nullptr, nullptr, t1_bf, Mrows, HIDd, Dd, 0);
    gemm_wmma<EPI_SCALE_ACC><<<gD, 256, 0, stream>>>(
        t1_bf, w2 + (size_t)e * HIDd * Dd, b2 + (size_t)e * Dd,
        nullptr, scl + e, core, nullptr, Mrows, Dd, HIDd, e > 0);
  }
  // 8) out = x1 + LN(h + core)
  ln_kernel<<<Mrows, 256, 0, stream>>>(h_f, core, mlg, mlb, nullptr, (float*)d_out, x1);
}
